// CausalSelfAttention_54881092108566
// MI455X (gfx1250) — matrix operand order repaired and checked by reading, not compiled, not run on hardware
//
#include <hip/hip_runtime.h>
#include <hip/hip_bf16.h>

typedef __attribute__((ext_vector_type(16))) _Float16 v16h;
typedef __attribute__((ext_vector_type(8)))  _Float16 v8h;
typedef __attribute__((ext_vector_type(4)))  _Float16 v4h;
typedef __attribute__((ext_vector_type(8)))  float    v8f;

union HFrag { v16h v; v8h h[2]; };

static __device__ __forceinline__ v8f wmma_f16(v16h a, v16h b, v8f c) {
  // D(f32 16x16) = A(f16 16x32) * B(f16 32x16) + C
  return __builtin_amdgcn_wmma_f32_16x16x32_f16(false, a, false, b, (short)0, c,
                                                false, false);
}

// CDNA5 async global->LDS copy, GVS mode: saddr(base) + 32-bit vgpr byte offset.
// Tracked by ASYNCcnt; paired with s_wait_asynccnt below.
static __device__ __forceinline__ void async_copy_b128(uint32_t lds_byte_addr,
                                                       const _Float16* base,
                                                       uint32_t byte_off) {
  asm volatile("global_load_async_to_lds_b128 %0, %1, %2"
               :: "v"(lds_byte_addr), "v"(byte_off), "s"(base)
               : "memory");
}
static __device__ __forceinline__ void wait_async_le4() {
  asm volatile("s_wait_asynccnt 0x4" ::: "memory");
}
static __device__ __forceinline__ void wait_async_0() {
  asm volatile("s_wait_asynccnt 0x0" ::: "memory");
}

constexpr int Bn = 4, T = 2048, C = 1024, H = 16, D = 64;
constexpr int M_TOT = Bn * T;    // 8192 tokens
constexpr int N_QKV = 3 * C;     // 3072

// ---------------------------------------------------------------------------
// One-shot precision conversion kernels (f32 -> f16), so both GEMMs stream
// pure-f16 tiles via the async path and every re-read costs half the HBM bytes.
// ---------------------------------------------------------------------------
__global__ __launch_bounds__(256) void cvt_f16_kernel(
    const float* __restrict__ in, _Float16* __restrict__ out, int n4)
{
  for (int i = blockIdx.x * blockDim.x + threadIdx.x; i < n4;
       i += gridDim.x * blockDim.x) {
    const float4 fv = *(const float4*)(in + (size_t)i * 4);
    v4h hv;
    hv.x = (_Float16)fv.x; hv.y = (_Float16)fv.y;
    hv.z = (_Float16)fv.z; hv.w = (_Float16)fv.w;
    *(v4h*)(out + (size_t)i * 4) = hv;
  }
}

// out[n*K + k] = (f16) in[k*N + n]  (transpose so B tiles are n-major,
// k-contiguous: B fragments become two contiguous ds_load_b128 per lane)
__global__ __launch_bounds__(256) void cvt_f16_t_kernel(
    const float* __restrict__ in, _Float16* __restrict__ out, int K, int N)
{
  const size_t total = (size_t)K * N;
  for (size_t i = blockIdx.x * (size_t)blockDim.x + threadIdx.x; i < total;
       i += (size_t)gridDim.x * blockDim.x) {
    const size_t k = i / N, n = i % N;
    out[n * (size_t)K + k] = (_Float16)in[i];
  }
}

// ---------------------------------------------------------------------------
// Kernel 1: qkv = xh @ WaT^T + b_attn.
// Workgroup tile 128(M) x 128(N), 8 waves as 4(M) x 2(N): each wave 32x64
// (2 A-frags x 4 B-frags = 8 WMMA per K-step). K step 32, K = 1024.
// Tiles stream global->LDS with global_load_async_to_lds_b128, double
// buffered on ASYNCcnt. Epilogue scatters f16 q[B,H,T,D], k[B,H,T,D],
// vT[B,H,D,T].
// ---------------------------------------------------------------------------
__global__ __launch_bounds__(256) void qkv_gemm_kernel(
    const _Float16* __restrict__ xh, const _Float16* __restrict__ WT,
    const float* __restrict__ bias,
    _Float16* __restrict__ qh, _Float16* __restrict__ kh,
    _Float16* __restrict__ vth)
{
  __shared__ _Float16 Abuf[2][128 * 32];   // (m, k) row-major
  __shared__ _Float16 Bbuf[2][128 * 32];   // (n, k) row-major (pre-transposed W)
  const int tid   = threadIdx.x;
  const int wave  = tid >> 5, lane = tid & 31;
  const int lhalf = lane >> 4, lq = lane & 15;
  const int waveM = wave & 3, waveN = wave >> 2;
  const int m0 = blockIdx.x * 128;
  const int n0 = blockIdx.y * 128;
  constexpr int NK = C / 32;

  // issue one K-tile (A: 128x32, B: 128x32) = 4 async b128 per thread
  auto issue = [&](int kt, int buf) {
#pragma unroll
    for (int i = 0; i < 2; ++i) {
      const int c = tid + i * 256;             // 512 b128 chunks
      const int row = c >> 2, c4 = c & 3;
      async_copy_b128((uint32_t)(uintptr_t)&Abuf[buf][row * 32 + c4 * 8],
                      xh, (uint32_t)(((m0 + row) * C + kt * 32 + c4 * 8) * 2));
    }
#pragma unroll
    for (int i = 0; i < 2; ++i) {
      const int c = tid + i * 256;
      const int row = c >> 2, c4 = c & 3;
      async_copy_b128((uint32_t)(uintptr_t)&Bbuf[buf][row * 32 + c4 * 8],
                      WT, (uint32_t)(((n0 + row) * C + kt * 32 + c4 * 8) * 2));
    }
  };

  v8f acc[2][4] = {};
  auto compute = [&](int buf) {
    HFrag a[2];
#pragma unroll
    for (int mi = 0; mi < 2; ++mi) {
      const int mrow = waveM * 32 + mi * 16 + lq;
      a[mi].h[0] = *(const v8h*)&Abuf[buf][mrow * 32 + 8 * lhalf];
      a[mi].h[1] = *(const v8h*)&Abuf[buf][mrow * 32 + 16 + 8 * lhalf];
    }
#pragma unroll
    for (int f = 0; f < 4; ++f) {
      HFrag b;
      const int nrow = waveN * 64 + f * 16 + lq;
      b.h[0] = *(const v8h*)&Bbuf[buf][nrow * 32 + 8 * lhalf];
      b.h[1] = *(const v8h*)&Bbuf[buf][nrow * 32 + 16 + 8 * lhalf];
#pragma unroll
      for (int mi = 0; mi < 2; ++mi)
        acc[mi][f] = wmma_f16(a[mi].v, b.v, acc[mi][f]);
    }
  };

  issue(0, 0);
  for (int kt = 0; kt < NK - 1; ++kt) {
    issue(kt + 1, (kt + 1) & 1);
    wait_async_le4();            // previous tile's 4 asyncs have landed
    __syncthreads();
    compute(kt & 1);
    __syncthreads();             // all waves done reading before refill
  }
  wait_async_0();
  __syncthreads();
  compute((NK - 1) & 1);

  // ---- epilogue: bias + scatter to q / k / vT (f16) ----
#pragma unroll
  for (int f = 0; f < 4; ++f) {
    const int n = n0 + waveN * 64 + f * 16 + lq;
    const float bv = bias[n];
#pragma unroll
    for (int mi = 0; mi < 2; ++mi) {
#pragma unroll
      for (int r = 0; r < 8; ++r) {
        const int row  = m0 + waveM * 32 + mi * 16 + 8 * lhalf + r;
        const int bidx = row >> 11;            // row / T
        const int tt   = row & (T - 1);
        const _Float16 hv = (_Float16)(acc[mi][f][r] + bv);
        if (n < C) {
          const int hh = n >> 6, d = n & 63;
          qh[(((size_t)(bidx * H + hh)) * T + tt) * D + d] = hv;
        } else if (n < 2 * C) {
          const int hd = n - C, hh = hd >> 6, d = hd & 63;
          kh[(((size_t)(bidx * H + hh)) * T + tt) * D + d] = hv;
        } else {
          const int hd = n - 2 * C, hh = hd >> 6, d = hd & 63;
          vth[(((size_t)(bidx * H + hh)) * D + d) * T + tt] = hv;
        }
      }
    }
  }
}

// ---------------------------------------------------------------------------
// Kernel 2: flash attention. One wave owns one 16-row Q block; key blocks of
// 32 with online softmax. 8 WMMAs per key block (4 for S, 4 for PV).
// ---------------------------------------------------------------------------
__global__ __launch_bounds__(256) void flash_attn_kernel(
    const _Float16* __restrict__ qh, const _Float16* __restrict__ kh,
    const _Float16* __restrict__ vth, _Float16* __restrict__ yh)
{
  __shared__ _Float16 P[8][16 * 32];   // per-wave P tile (16 q rows x 32 keys)
  const int tid   = threadIdx.x;
  const int wave  = tid >> 5, lane = tid & 31;
  const int lhalf = lane >> 4, lq = lane & 15;
  const int g     = blockIdx.x * 8 + wave;   // global wave id
  const int bh    = g >> 7;                  // (b*H + h), T/16 = 128 qblocks
  const int qbase = (g & 127) * 16;

  const _Float16* qp = qh  + (size_t)bh * T * D;
  const _Float16* kp = kh  + (size_t)bh * T * D;
  const _Float16* vp = vth + (size_t)bh * D * T;

  HFrag aq[2];
  {
    const _Float16* qr = qp + (size_t)(qbase + lq) * D;
#pragma unroll
    for (int kk = 0; kk < 2; ++kk) {
      aq[kk].h[0] = *(const v8h*)(qr + kk * 32 + 8 * lhalf);
      aq[kk].h[1] = *(const v8h*)(qr + kk * 32 + 16 + 8 * lhalf);
    }
  }

  float mrow[8], lrow[8];
  v8f acc[4] = {};
#pragma unroll
  for (int r = 0; r < 8; ++r) { mrow[r] = -1e30f; lrow[r] = 0.0f; }
  const float scale = 0.125f;   // 1/sqrt(64)

  const int nkb = (qbase + 16 + 31) >> 5;
  for (int jbI = 0; jbI < nkb; ++jbI) {
    const int jb = jbI * 32;
    v8f s[2] = {};
#pragma unroll
    for (int nh = 0; nh < 2; ++nh) {
#pragma unroll
      for (int kk = 0; kk < 2; ++kk) {
        HFrag bk;   // B[k=d, n=key]: lane col = key row (lq), contiguous d
        const _Float16* kb =
            kp + (size_t)(jb + nh * 16 + lq) * D + kk * 32 + 8 * lhalf;
        bk.h[0] = *(const v8h*)kb;
        bk.h[1] = *(const v8h*)(kb + 16);
        s[nh] = wmma_f16(aq[kk].v, bk.v, s[nh]);
      }
    }
    const int qi_base = qbase + 8 * lhalf;
#pragma unroll
    for (int r = 0; r < 8; ++r) {
      const int qi = qi_base + r;
      float s0 = s[0][r] * scale;
      float s1 = s[1][r] * scale;
      if (jb + lq      > qi) s0 = -1e30f;   // causal mask
      if (jb + 16 + lq > qi) s1 = -1e30f;
      float mx = fmaxf(s0, s1);
#pragma unroll
      for (int off = 8; off >= 1; off >>= 1)
        mx = fmaxf(mx, __shfl_xor(mx, off, 16));
      const float mn   = fmaxf(mrow[r], mx);
      const float corr = __expf(mrow[r] - mn);
      const float p0   = __expf(s0 - mn);
      const float p1   = __expf(s1 - mn);
      float rs = p0 + p1;
#pragma unroll
      for (int off = 8; off >= 1; off >>= 1)
        rs += __shfl_xor(rs, off, 16);
      lrow[r] = lrow[r] * corr + rs;
      mrow[r] = mn;
#pragma unroll
      for (int dc = 0; dc < 4; ++dc) acc[dc][r] = acc[dc][r] * corr;
      _Float16* prow = &P[wave][(8 * lhalf + r) * 32];
      prow[lq]      = (_Float16)p0;
      prow[16 + lq] = (_Float16)p1;
    }
    asm volatile("s_wait_dscnt 0" ::: "memory");   // cross-lane P visibility
    HFrag pa;
    pa.h[0] = *(const v8h*)&P[wave][lq * 32 + 8 * lhalf];
    pa.h[1] = *(const v8h*)&P[wave][lq * 32 + 16 + 8 * lhalf];
    asm volatile("s_wait_dscnt 0" ::: "memory");   // reads done before overwrite
#pragma unroll
    for (int dc = 0; dc < 4; ++dc) {
      HFrag bv;
      const _Float16* vb = vp + (size_t)(dc * 16 + lq) * T + jb + 8 * lhalf;
      bv.h[0] = *(const v8h*)vb;
      bv.h[1] = *(const v8h*)(vb + 16);
      acc[dc] = wmma_f16(pa.v, bv.v, acc[dc]);
    }
  }

  const int bidx = bh >> 4, hh = bh & 15;
#pragma unroll
  for (int r = 0; r < 8; ++r) {
    const int t = qbase + 8 * lhalf + r;
    const float inv = 1.0f / lrow[r];
#pragma unroll
    for (int dc = 0; dc < 4; ++dc) {
      yh[((size_t)(bidx * T + t)) * C + hh * D + dc * 16 + lq] =
          (_Float16)(acc[dc][r] * inv);
    }
  }
}

// ---------------------------------------------------------------------------
// Kernel 3: out = yh @ WpT^T + b_proj (f32 output). Same async double-buffered
// 128x128 WMMA tile as kernel 1.
// ---------------------------------------------------------------------------
__global__ __launch_bounds__(256) void proj_gemm_kernel(
    const _Float16* __restrict__ yh, const _Float16* __restrict__ WT,
    const float* __restrict__ bias, float* __restrict__ out)
{
  __shared__ _Float16 Abuf[2][128 * 32];
  __shared__ _Float16 Bbuf[2][128 * 32];
  const int tid   = threadIdx.x;
  const int wave  = tid >> 5, lane = tid & 31;
  const int lhalf = lane >> 4, lq = lane & 15;
  const int waveM = wave & 3, waveN = wave >> 2;
  const int m0 = blockIdx.x * 128;
  const int n0 = blockIdx.y * 128;
  constexpr int NK = C / 32;

  auto issue = [&](int kt, int buf) {
#pragma unroll
    for (int i = 0; i < 2; ++i) {
      const int c = tid + i * 256;
      const int row = c >> 2, c4 = c & 3;
      async_copy_b128((uint32_t)(uintptr_t)&Abuf[buf][row * 32 + c4 * 8],
                      yh, (uint32_t)(((m0 + row) * C + kt * 32 + c4 * 8) * 2));
    }
#pragma unroll
    for (int i = 0; i < 2; ++i) {
      const int c = tid + i * 256;
      const int row = c >> 2, c4 = c & 3;
      async_copy_b128((uint32_t)(uintptr_t)&Bbuf[buf][row * 32 + c4 * 8],
                      WT, (uint32_t)(((n0 + row) * C + kt * 32 + c4 * 8) * 2));
    }
  };

  v8f acc[2][4] = {};
  auto compute = [&](int buf) {
    HFrag a[2];
#pragma unroll
    for (int mi = 0; mi < 2; ++mi) {
      const int mrow = waveM * 32 + mi * 16 + lq;
      a[mi].h[0] = *(const v8h*)&Abuf[buf][mrow * 32 + 8 * lhalf];
      a[mi].h[1] = *(const v8h*)&Abuf[buf][mrow * 32 + 16 + 8 * lhalf];
    }
#pragma unroll
    for (int f = 0; f < 4; ++f) {
      HFrag b;
      const int nrow = waveN * 64 + f * 16 + lq;
      b.h[0] = *(const v8h*)&Bbuf[buf][nrow * 32 + 8 * lhalf];
      b.h[1] = *(const v8h*)&Bbuf[buf][nrow * 32 + 16 + 8 * lhalf];
#pragma unroll
      for (int mi = 0; mi < 2; ++mi)
        acc[mi][f] = wmma_f16(a[mi].v, b.v, acc[mi][f]);
    }
  };

  issue(0, 0);
  for (int kt = 0; kt < NK - 1; ++kt) {
    issue(kt + 1, (kt + 1) & 1);
    wait_async_le4();
    __syncthreads();
    compute(kt & 1);
    __syncthreads();
  }
  wait_async_0();
  __syncthreads();
  compute((NK - 1) & 1);

#pragma unroll
  for (int f = 0; f < 4; ++f) {
    const int n = n0 + waveN * 64 + f * 16 + lq;
    const float bv = bias[n];
#pragma unroll
    for (int mi = 0; mi < 2; ++mi) {
#pragma unroll
      for (int r = 0; r < 8; ++r) {
        const int row = m0 + waveM * 32 + mi * 16 + 8 * lhalf + r;
        out[(size_t)row * C + n] = acc[mi][f][r] + bv;
      }
    }
  }
}

// ---------------------------------------------------------------------------
extern "C" void kernel_launch(void* const* d_in, const int* in_sizes, int n_in,
                              void* d_out, int out_size, void* d_ws, size_t ws_size,
                              hipStream_t stream) {
  const float* x      = (const float*)d_in[0];
  const float* W_attn = (const float*)d_in[1];
  const float* b_attn = (const float*)d_in[2];
  const float* W_proj = (const float*)d_in[3];
  const float* b_proj = (const float*)d_in[4];
  float* out = (float*)d_out;

  const size_t XS = (size_t)M_TOT * C;        // 8,388,608
  const size_t WA = (size_t)C * N_QKV;        // 3,145,728
  const size_t WP = (size_t)C * C;            // 1,048,576
  const size_t QS = (size_t)Bn * H * T * D;   // 8,388,608

  _Float16* xh  = (_Float16*)d_ws;            // x   [B*T, C]       f16
  _Float16* WaT = xh  + XS;                   // W_attn^T [3C, C]   f16
  _Float16* WpT = WaT + WA;                   // W_proj^T [C, C]    f16
  _Float16* qh  = WpT + WP;                   // q  [B,H,T,D]
  _Float16* kh  = qh  + QS;                   // k  [B,H,T,D]
  _Float16* vth = kh  + QS;                   // vT [B,H,D,T]
  _Float16* yh  = vth + QS;                   // y  [B,T,C]

  cvt_f16_kernel<<<2048, 256, 0, stream>>>(x, xh, (int)(XS / 4));
  cvt_f16_t_kernel<<<(int)(WA / 256), 256, 0, stream>>>(W_attn, WaT, C, N_QKV);
  cvt_f16_t_kernel<<<(int)(WP / 256), 256, 0, stream>>>(W_proj, WpT, C, C);

  dim3 g1(M_TOT / 128, N_QKV / 128);          // 64 x 24
  qkv_gemm_kernel<<<g1, 256, 0, stream>>>(xh, WaT, b_attn, qh, kh, vth);

  const int attn_blocks = (Bn * H * (T / 16)) / 8;   // 1024 WGs, 8 waves each
  flash_attn_kernel<<<attn_blocks, 256, 0, stream>>>(qh, kh, vth, yh);

  dim3 g3(M_TOT / 128, C / 128);              // 64 x 8
  proj_gemm_kernel<<<g3, 256, 0, stream>>>(yh, WpT, b_proj, out);
}
